// TrajEncoder_62319975465181
// MI455X (gfx1250) — matrix operand order repaired and checked by reading, compiled, not run on hardware
//
#include <hip/hip_runtime.h>
#include <math.h>

typedef __attribute__((ext_vector_type(16))) _Float16 v16h;
typedef __attribute__((ext_vector_type(8)))  _Float16 v8h;
typedef __attribute__((ext_vector_type(8)))  float    v8f;

#define B_    8192
#define TC_   4
#define L_    15
#define V_    128
#define KSEL  5
#define HID_  1024
#define OUT_  512
#define INF_  45
#define KPAD  64
#define M_    (B_*KSEL)      // 40960 rows, multiple of 32

// ---------------------------------------------------------------------------
// Kernel 1: relative trajectory. poses (B,19,4,4) f32 -> traj (B,15,3) f32
// ---------------------------------------------------------------------------
__global__ void traj_kernel(const float* __restrict__ poses,
                            float* __restrict__ traj) {
  int b = blockIdx.x * blockDim.x + threadIdx.x;
  if (b >= B_) return;
  const float* p3 = poses + ((size_t)b * 19 + 3) * 16;
  float c3 = p3[0];   // R[0][0]
  float s3 = p3[4];   // R[1][0]
  float tx3 = p3[3], ty3 = p3[7];
  float* out = traj + (size_t)b * 45;
  for (int g = 0; g < L_; ++g) {
    const float* pj = poses + ((size_t)b * 19 + 4 + g) * 16;
    float cj = pj[0], sj = pj[4];
    float dx = pj[3] - tx3, dy = pj[7] - ty3;
    float rx =  c3 * dx + s3 * dy;
    float ry = -s3 * dx + c3 * dy;
    float yw = atan2f(c3 * sj - s3 * cj, c3 * cj + s3 * sj);
    out[g * 3 + 0] = rx;
    out[g * 3 + 1] = ry;
    out[g * 3 + 2] = yw;
  }
}

// ---------------------------------------------------------------------------
// Kernel 2: top-K nearest vocab (stable ascending) + build f16 X matrices.
// ---------------------------------------------------------------------------
__global__ void build_x_kernel(const float* __restrict__ vocab,
                               const float* __restrict__ traj,
                               _Float16* __restrict__ Xv,
                               _Float16* __restrict__ Xo) {
  int b = blockIdx.x * blockDim.x + threadIdx.x;
  if (b >= B_) return;
  const float* tb = traj + (size_t)b * 45;
  float lx = tb[42], ly = tb[43], lh = tb[44];
  int chosen[KSEL];
  for (int k = 0; k < KSEL; ++k) {
    float best = 3.4e38f;
    int bi = 0;
    for (int v = 0; v < V_; ++v) {
      bool used = false;
      for (int kk = 0; kk < k; ++kk) used = used || (chosen[kk] == v);
      if (used) continue;
      const float* vl = vocab + (size_t)v * 45 + 42;
      float d0 = lx - vl[0], d1 = ly - vl[1], d2 = lh - vl[2];
      float d = d0 * d0 + d1 * d1 + d2 * d2;
      if (d < best) { best = d; bi = v; }   // strict < => lowest index wins ties
    }
    chosen[k] = bi;
    size_t m = (size_t)b * KSEL + k;
    const float* vb = vocab + (size_t)bi * 45;
    _Float16* xv = Xv + m * KPAD;
    _Float16* xo = Xo + m * KPAD;
    for (int l = 0; l < L_; ++l) {
      float t0 = vb[l * 3 + 0], t1 = vb[l * 3 + 1], t2 = vb[l * 3 + 2];
      xv[l * 3 + 0] = (_Float16)(2.f * (t0 + 1.57f)  / 66.74f - 1.f);
      xv[l * 3 + 1] = (_Float16)(2.f * (t1 + 19.68f) / 42.f   - 1.f);
      xv[l * 3 + 2] = (_Float16)(2.f * (t2 + 1.67f)  / 3.53f  - 1.f);
      xo[l * 3 + 0] = (_Float16)(tb[l * 3 + 0] - t0);
      xo[l * 3 + 1] = (_Float16)(tb[l * 3 + 1] - t1);
      xo[l * 3 + 2] = (_Float16)(tb[l * 3 + 2] - t2);
    }
    for (int j = INF_; j < KPAD; ++j) { xv[j] = (_Float16)0.f; xo[j] = (_Float16)0.f; }
  }
}

// ---------------------------------------------------------------------------
// Weight packing into WMMA B-fragment layout:
//   Bp[((kb*NCOL + col)*2 + half)*16 + j] = W[kb*32 + half*16 + j][col]
// ---------------------------------------------------------------------------
__global__ void pack_w1_kernel(const float* __restrict__ W1,
                               _Float16* __restrict__ W1p) {
  int t = blockIdx.x * blockDim.x + threadIdx.x;
  if (t >= 2 * HID_ * 32) return;
  int j    = t & 15;
  int half = (t >> 4) & 1;
  int rest = t >> 5;
  int col  = rest & (HID_ - 1);
  int kb   = rest >> 10;
  int krow = kb * 32 + half * 16 + j;
  float v = (krow < INF_) ? W1[(size_t)krow * HID_ + col] : 0.f;
  W1p[t] = (_Float16)v;
}

__global__ void pack_w2_kernel(const float* __restrict__ W2,
                               _Float16* __restrict__ W2p) {
  int t = blockIdx.x * blockDim.x + threadIdx.x;
  if (t >= 32 * OUT_ * 32) return;
  int j    = t & 15;
  int half = (t >> 4) & 1;
  int rest = t >> 5;
  int col  = rest & (OUT_ - 1);
  int kb   = rest >> 9;
  int krow = kb * 32 + half * 16 + j;
  W2p[t] = (_Float16)W2[(size_t)krow * OUT_ + col];
}

// ---------------------------------------------------------------------------
// Stage 1: H = LN(gelu(X @ W1 + b1))*g + be  -> f16 H (M x 1024)
// Block = 16 rows x 1024 cols; 16 waves; each wave: 4 C-tiles (64 cols).
// ---------------------------------------------------------------------------
__global__ void __launch_bounds__(512) stage1_kernel(
    const _Float16* __restrict__ X, const _Float16* __restrict__ W1p,
    const float* __restrict__ b1, const float* __restrict__ g,
    const float* __restrict__ be, _Float16* __restrict__ H) {
  __shared__ float sSum[16][16];
  __shared__ float sSq[16][16];
  __shared__ float sMean[16];
  __shared__ float sInv[16];

  int tid  = threadIdx.x;
  int wave = tid >> 5;                // 0..15
  int lane = tid & 31;
  int half = lane >> 4;
  int lmod = lane & 15;
  size_t rowBase = (size_t)blockIdx.x * 16;
  int colBase = wave * 64;

  const _Float16* xBase = X + (rowBase + lmod) * KPAD + half * 16;
  const _Float16* wBase[4];
#pragma unroll
  for (int t = 0; t < 4; ++t)
    wBase[t] = W1p + (((size_t)(colBase + t * 16 + lmod)) * 2 + half) * 16;

  v8f acc[4] = {};
#pragma unroll
  for (int kb = 0; kb < 2; ++kb) {
    v8h alo = *(const v8h*)(xBase + kb * 32);
    v8h ahi = *(const v8h*)(xBase + kb * 32 + 8);
    v16h a = __builtin_shufflevector(alo, ahi, 0,1,2,3,4,5,6,7,8,9,10,11,12,13,14,15);
#pragma unroll
    for (int t = 0; t < 4; ++t) {
      const _Float16* wp = wBase[t] + (size_t)kb * HID_ * 32;
      v8h blo = *(const v8h*)wp;
      v8h bhi = *(const v8h*)(wp + 8);
      v16h bf = __builtin_shufflevector(blo, bhi, 0,1,2,3,4,5,6,7,8,9,10,11,12,13,14,15);
      acc[t] = __builtin_amdgcn_wmma_f32_16x16x32_f16(
          false, a, false, bf, (short)0, acc[t], false, false);
    }
  }

  // Exact GeLU in-place + per-lane partial LN stats (row of acc[i] = i+8*half).
  float s[8], q[8];
#pragma unroll
  for (int i = 0; i < 8; ++i) { s[i] = 0.f; q[i] = 0.f; }
#pragma unroll
  for (int t = 0; t < 4; ++t) {
    int col = colBase + t * 16 + lmod;
    float bcol = b1[col];
#pragma unroll
    for (int i = 0; i < 8; ++i) {
      float x = acc[t][i] + bcol;
      float gl = 0.5f * x * (1.f + erff(x * 0.70710678118654752f));
      acc[t][i] = gl;
      s[i] += gl;
      q[i] += gl * gl;
    }
  }
  for (int m = 1; m < 16; m <<= 1) {
#pragma unroll
    for (int i = 0; i < 8; ++i) {
      s[i] += __shfl_xor(s[i], m, 32);
      q[i] += __shfl_xor(q[i], m, 32);
    }
  }
  if (lmod == 0) {
#pragma unroll
    for (int i = 0; i < 8; ++i) {
      sSum[i + 8 * half][wave] = s[i];
      sSq [i + 8 * half][wave] = q[i];
    }
  }
  __syncthreads();
  if (tid < 16) {
    float su = 0.f, sq = 0.f;
#pragma unroll
    for (int w = 0; w < 16; ++w) { su += sSum[tid][w]; sq += sSq[tid][w]; }
    float mean = su * (1.f / 1024.f);
    float var  = sq * (1.f / 1024.f) - mean * mean;
    sMean[tid] = mean;
    sInv[tid]  = rsqrtf(var + 1e-5f);
  }
  __syncthreads();
#pragma unroll
  for (int t = 0; t < 4; ++t) {
    int col = colBase + t * 16 + lmod;
    float gg = g[col], bb = be[col];
#pragma unroll
    for (int i = 0; i < 8; ++i) {
      int row = i + 8 * half;
      float v = (acc[t][i] - sMean[row]) * sInv[row] * gg + bb;
      H[(rowBase + row) * HID_ + col] = (_Float16)v;
    }
  }
}

// ---------------------------------------------------------------------------
// Stage 2: Y(+)= H @ W2 + b2.
// Block = 32 rows x 512 cols; 8 waves; each wave: 2 row-tiles x 4 col-tiles.
// Each B fragment reused by 2 WMMAs; each A fragment by 4.
// ---------------------------------------------------------------------------
__global__ void __launch_bounds__(256) stage2_kernel(
    const _Float16* __restrict__ H, const _Float16* __restrict__ W2p,
    const float* __restrict__ b2, float* __restrict__ Y, int accumulate) {
  int tid  = threadIdx.x;
  int wave = tid >> 5;                // 0..7
  int lane = tid & 31;
  int half = lane >> 4;
  int lmod = lane & 15;
  size_t rowBase = (size_t)blockIdx.x * 32;
  int colBase = wave * 64;

  const _Float16* hBase0 = H + (rowBase + lmod) * HID_ + half * 16;
  const _Float16* hBase1 = hBase0 + (size_t)16 * HID_;
  const _Float16* wBase[4];
#pragma unroll
  for (int t = 0; t < 4; ++t)
    wBase[t] = W2p + (((size_t)(colBase + t * 16 + lmod)) * 2 + half) * 16;

  v8f acc0[4] = {};
  v8f acc1[4] = {};
#pragma unroll
  for (int kb = 0; kb < 32; ++kb) {
    v8h a0lo = *(const v8h*)(hBase0 + kb * 32);
    v8h a0hi = *(const v8h*)(hBase0 + kb * 32 + 8);
    v16h a0 = __builtin_shufflevector(a0lo, a0hi, 0,1,2,3,4,5,6,7,8,9,10,11,12,13,14,15);
    v8h a1lo = *(const v8h*)(hBase1 + kb * 32);
    v8h a1hi = *(const v8h*)(hBase1 + kb * 32 + 8);
    v16h a1 = __builtin_shufflevector(a1lo, a1hi, 0,1,2,3,4,5,6,7,8,9,10,11,12,13,14,15);
#pragma unroll
    for (int t = 0; t < 4; ++t) {
      const _Float16* wp = wBase[t] + (size_t)kb * OUT_ * 32;
      v8h blo = *(const v8h*)wp;
      v8h bhi = *(const v8h*)(wp + 8);
      v16h bf = __builtin_shufflevector(blo, bhi, 0,1,2,3,4,5,6,7,8,9,10,11,12,13,14,15);
      acc0[t] = __builtin_amdgcn_wmma_f32_16x16x32_f16(
          false, a0, false, bf, (short)0, acc0[t], false, false);
      acc1[t] = __builtin_amdgcn_wmma_f32_16x16x32_f16(
          false, a1, false, bf, (short)0, acc1[t], false, false);
    }
  }

#pragma unroll
  for (int t = 0; t < 4; ++t) {
    int col = colBase + t * 16 + lmod;
    float bias = b2[col];
#pragma unroll
    for (int i = 0; i < 8; ++i) {
      size_t idx0 = (rowBase + i + 8 * half) * OUT_ + col;
      size_t idx1 = idx0 + (size_t)16 * OUT_;
      float v0 = acc0[t][i] + bias;
      float v1 = acc1[t][i] + bias;
      if (accumulate) { v0 += Y[idx0]; v1 += Y[idx1]; }
      Y[idx0] = v0;
      Y[idx1] = v1;
    }
  }
}

// ---------------------------------------------------------------------------
extern "C" void kernel_launch(void* const* d_in, const int* in_sizes, int n_in,
                              void* d_out, int out_size, void* d_ws, size_t ws_size,
                              hipStream_t stream) {
  const float* poses = (const float*)d_in[0];
  const float* vocab = (const float*)d_in[1];
  const float* w1v = (const float*)d_in[2];
  const float* b1v = (const float*)d_in[3];
  const float* gv  = (const float*)d_in[4];
  const float* bev = (const float*)d_in[5];
  const float* w2v = (const float*)d_in[6];
  const float* b2v = (const float*)d_in[7];
  const float* w1o = (const float*)d_in[8];
  const float* b1o = (const float*)d_in[9];
  const float* go  = (const float*)d_in[10];
  const float* beo = (const float*)d_in[11];
  const float* w2o = (const float*)d_in[12];
  const float* b2o = (const float*)d_in[13];
  float* Y = (float*)d_out;

  char* ws = (char*)d_ws;
  size_t off = 0;
  auto carve = [&](size_t bytes) -> void* {
    void* p = ws + off;
    off += (bytes + 255) & ~(size_t)255;
    return p;
  };
  float*    traj = (float*)   carve((size_t)B_ * 45 * sizeof(float));
  _Float16* Xv   = (_Float16*)carve((size_t)M_ * KPAD * sizeof(_Float16));
  _Float16* Xo   = (_Float16*)carve((size_t)M_ * KPAD * sizeof(_Float16));
  _Float16* W1vp = (_Float16*)carve((size_t)2 * HID_ * 32 * sizeof(_Float16));
  _Float16* W1op = (_Float16*)carve((size_t)2 * HID_ * 32 * sizeof(_Float16));
  _Float16* W2vp = (_Float16*)carve((size_t)32 * OUT_ * 32 * sizeof(_Float16));
  _Float16* W2op = (_Float16*)carve((size_t)32 * OUT_ * 32 * sizeof(_Float16));
  _Float16* Hbuf = (_Float16*)carve((size_t)M_ * HID_ * sizeof(_Float16));
  (void)ws_size; (void)in_sizes; (void)n_in; (void)out_size;

  // Prep
  traj_kernel   <<<B_ / 256, 256, 0, stream>>>(poses, traj);
  build_x_kernel<<<B_ / 256, 256, 0, stream>>>(vocab, traj, Xv, Xo);
  pack_w1_kernel<<<(2 * HID_ * 32) / 256, 256, 0, stream>>>(w1v, W1vp);
  pack_w1_kernel<<<(2 * HID_ * 32) / 256, 256, 0, stream>>>(w1o, W1op);
  pack_w2_kernel<<<(32 * OUT_ * 32) / 256, 256, 0, stream>>>(w2v, W2vp);
  pack_w2_kernel<<<(32 * OUT_ * 32) / 256, 256, 0, stream>>>(w2o, W2op);

  int s1_blocks = M_ / 16;   // 2560, 512 threads each
  int s2_blocks = M_ / 32;   // 1280, 256 threads each

  // Vocab branch: write
  stage1_kernel<<<s1_blocks, 512, 0, stream>>>(Xv, W1vp, b1v, gv, bev, Hbuf);
  stage2_kernel<<<s2_blocks, 256, 0, stream>>>(Hbuf, W2vp, b2v, Y, 0);
  // Offset branch: accumulate (reuses H buffer)
  stage1_kernel<<<s1_blocks, 512, 0, stream>>>(Xo, W1op, b1o, go, beo, Hbuf);
  stage2_kernel<<<s2_blocks, 256, 0, stream>>>(Hbuf, W2op, b2o, Y, 1);
}
